// TnoFD_43009802502425
// MI455X (gfx1250) — matrix operand order repaired and checked by reading, not compiled, not run on hardware
//
#include <hip/hip_runtime.h>

#define HH 8
#define DD 64
#define NSEQ 4096
#define BB 8
#define NF 4097
#define PI_F 3.14159265358979323846f

typedef __attribute__((ext_vector_type(16))) __bf16 v16bf;
typedef __attribute__((ext_vector_type(16))) unsigned short v16u;
typedef __attribute__((ext_vector_type(8)))  float v8f;

struct c2 { float x, y; };
struct alignas(16) f4 { c2 s0, s1; };          // two sequences interleaved
__device__ inline c2 cmul(c2 a, c2 b){ return {a.x*b.x - a.y*b.y, a.x*b.y + a.y*b.x}; }

__device__ inline unsigned short f2bf(float f){
  unsigned int u = __float_as_uint(f);
  u += 0x7FFFu + ((u >> 16) & 1u);            // round-to-nearest-even
  return (unsigned short)(u >> 16);
}

// ---------------------------------------------------------------------------
// Kernel 1: RPE MLP -> complex filter spectrum filt[h][f][0:64]=re, [64:128]=im
// One wave per 16-row tile; GEMMs via v_wmma_f32_16x16x32_bf16 (f32 accum).
// ---------------------------------------------------------------------------
__global__ void __launch_bounds__(32)
rpe_mlp_kernel(const float* __restrict__ w_in, const float* __restrict__ b_in,
               const float* __restrict__ w_layers, const float* __restrict__ b_layers,
               const float* __restrict__ w_out, const float* __restrict__ b_out,
               float* __restrict__ filt)
{
  __shared__ float act[16][65];
  __shared__ float nrm[16][65];
  const int tid = threadIdx.x;
  const int m  = tid & 15;
  const int hi = tid >> 4;
  const int r0 = blockIdx.x * 16;

  for(int idx = tid; idx < 16*64; idx += 32){
    int r = idx >> 6, j = idx & 63;
    float pos = PI_F * (float)(r0 + r) * (1.0f/4096.0f);
    act[r][j] = pos * w_in[j] + b_in[j];
  }
  __syncthreads();

  for(int layer = 0; layer < 3; ++layer){
    if(tid < 16){
      float s = 0.f;
      for(int j = 0; j < 64; ++j){ float v = act[tid][j]; s += v*v; }
      float inv = 1.0f / (sqrtf(s * (1.0f/64.0f)) + 1e-8f);
      for(int j = 0; j < 64; ++j){
        float v = act[tid][j] * inv;
        nrm[tid][j] = v > 0.f ? v : 0.f;
      }
    }
    __syncthreads();
    const float* W  = w_layers + layer*64*64;
    const float* bb = b_layers + layer*64;
    for(int nt = 0; nt < 4; ++nt){
      v8f acc = {};
      for(int kk = 0; kk < 2; ++kk){
        v16u au, bu;
        for(int v = 0; v < 8; ++v){
          int kb = (v < 4) ? (2*v + 8*hi) : (16 + 2*(v-4) + 8*hi);
          int k0 = kk*32 + kb;
          au[2*v]   = f2bf(nrm[m][k0]);
          au[2*v+1] = f2bf(nrm[m][k0+1]);
          int kw = kk*32 + kb;
          bu[2*v]   = f2bf(W[kw*64     + nt*16 + m]);
          bu[2*v+1] = f2bf(W[(kw+1)*64 + nt*16 + m]);
        }
        v16bf a    = __builtin_bit_cast(v16bf, au);
        v16bf bmat = __builtin_bit_cast(v16bf, bu);
        acc = __builtin_amdgcn_wmma_f32_16x16x32_bf16(false, a, false, bmat,
                                                      (short)0, acc, false, false);
      }
      for(int v = 0; v < 8; ++v){
        int row = v + 8*hi;
        int col = nt*16 + m;
        act[row][col] = acc[v] + bb[col];
      }
    }
    __syncthreads();
  }

  if(tid < 16){
    float s = 0.f;
    for(int j = 0; j < 64; ++j){ float v = act[tid][j]; s += v*v; }
    float inv = 1.0f / (sqrtf(s * (1.0f/64.0f)) + 1e-8f);
    for(int j = 0; j < 64; ++j){
      float v = act[tid][j] * inv;
      nrm[tid][j] = v > 0.f ? v : 0.f;
    }
  }
  __syncthreads();

  for(int nt = 0; nt < 64; ++nt){
    v8f acc = {};
    for(int kk = 0; kk < 2; ++kk){
      v16u au, bu;
      for(int v = 0; v < 8; ++v){
        int kb = (v < 4) ? (2*v + 8*hi) : (16 + 2*(v-4) + 8*hi);
        int k0 = kk*32 + kb;
        au[2*v]   = f2bf(nrm[m][k0]);
        au[2*v+1] = f2bf(nrm[m][k0+1]);
        int kw = kk*32 + kb;
        bu[2*v]   = f2bf(w_out[kw*1024     + nt*16 + m]);
        bu[2*v+1] = f2bf(w_out[(kw+1)*1024 + nt*16 + m]);
      }
      v16bf a    = __builtin_bit_cast(v16bf, au);
      v16bf bmat = __builtin_bit_cast(v16bf, bu);
      acc = __builtin_amdgcn_wmma_f32_16x16x32_bf16(false, a, false, bmat,
                                                    (short)0, acc, false, false);
    }
    int col = nt*16 + m;
    int hh  = col >> 7;
    int rem = col & 127;
    int im  = rem >= 64 ? 1 : 0;
    int d   = rem & 63;
    float bo = b_out[col];
    for(int v = 0; v < 8; ++v){
      int f = r0 + v + 8*hi;
      if(f < NF){
        float val = acc[v] + bo;
        if(im && (f == 0 || f == NF-1)) val = 0.f;
        filt[((size_t)hh*NF + f)*128 + (im ? 64 : 0) + d] = val;
      }
    }
  }
}

// ---------------------------------------------------------------------------
// DFT-8 butterfly, sign folded at compile time (SGN = -1 fwd, +1 inv).
// ---------------------------------------------------------------------------
template<int SGN>
__device__ inline void dft8(c2* a)
{
  const float S  = (float)SGN;
  const float c8 = 0.70710678118654752440f;
  c2 e0=a[0], e1=a[2], e2=a[4], e3=a[6];
  c2 o0=a[1], o1=a[3], o2=a[5], o3=a[7];
  c2 t0={e0.x+e2.x,e0.y+e2.y}, t1={e0.x-e2.x,e0.y-e2.y};
  c2 t2={e1.x+e3.x,e1.y+e3.y}, t3={e1.x-e3.x,e1.y-e3.y};
  c2 E0={t0.x+t2.x,t0.y+t2.y}, E2={t0.x-t2.x,t0.y-t2.y};
  c2 E1={t1.x - S*t3.y, t1.y + S*t3.x};
  c2 E3={t1.x + S*t3.y, t1.y - S*t3.x};
  c2 u0={o0.x+o2.x,o0.y+o2.y}, u1={o0.x-o2.x,o0.y-o2.y};
  c2 u2={o1.x+o3.x,o1.y+o3.y}, u3={o1.x-o3.x,o1.y-o3.y};
  c2 O0={u0.x+u2.x,u0.y+u2.y}, O2={u0.x-u2.x,u0.y-u2.y};
  c2 O1={u1.x - S*u3.y, u1.y + S*u3.x};
  c2 O3={u1.x + S*u3.y, u1.y - S*u3.x};
  c2 W1O1 = { c8*(O1.x - S*O1.y),  c8*(O1.y + S*O1.x) };   // w8^1 * O1
  c2 W2O2 = { -S*O2.y,             S*O2.x };               // w8^2 * O2
  c2 W3O3 = { c8*(-O3.x - S*O3.y), c8*(-O3.y + S*O3.x) };  // w8^3 * O3
  a[0]={E0.x+O0.x,  E0.y+O0.y};   a[4]={E0.x-O0.x,  E0.y-O0.y};
  a[1]={E1.x+W1O1.x,E1.y+W1O1.y}; a[5]={E1.x-W1O1.x,E1.y-W1O1.y};
  a[2]={E2.x+W2O2.x,E2.y+W2O2.y}; a[6]={E2.x-W2O2.x,E2.y-W2O2.y};
  a[3]={E3.x+W3O3.x,E3.y+W3O3.y}; a[7]={E3.x-W3O3.x,E3.y-W3O3.y};
}

// ---------------------------------------------------------------------------
// Radix-8 Stockham autosort FFT, N=4096 (8^4 -> 4 stages), 2 sequences per
// element (f4), twiddles shared. Result lands back in `src` (4 swaps).
// ---------------------------------------------------------------------------
template<int SGN>
__device__ f4* fft4096_r8(f4* src, f4* dst, int tid)
{
  int L = 1;
  for(int st = 0; st < 4; ++st){
    __syncthreads();
    const int sh = 3*st;
    for(int j = tid; j < 512; j += 256){
      int k   = j & (L - 1);
      int blk = j >> sh;
      f4 a[8];
      #pragma unroll
      for(int q = 0; q < 8; ++q) a[q] = src[j + q*512];
      float ang = (float)SGN * (0.25f*PI_F) * (float)k / (float)L;  // 2*pi*k/(8L)
      float sw, cw; __sincosf(ang, &sw, &cw);
      c2 w1={cw,sw};
      c2 w2=cmul(w1,w1), w3=cmul(w2,w1), w4=cmul(w2,w2);
      c2 w5=cmul(w4,w1), w6=cmul(w3,w3), w7=cmul(w4,w3);
      c2 w[8] = { {1.f,0.f}, w1, w2, w3, w4, w5, w6, w7 };
      #pragma unroll
      for(int q = 1; q < 8; ++q){
        a[q].s0 = cmul(a[q].s0, w[q]);
        a[q].s1 = cmul(a[q].s1, w[q]);
      }
      c2 p[8], r[8];
      #pragma unroll
      for(int q = 0; q < 8; ++q){ p[q] = a[q].s0; r[q] = a[q].s1; }
      dft8<SGN>(p);
      dft8<SGN>(r);
      int base = (blk << (sh + 3)) + k;       // blk*8L + k
      #pragma unroll
      for(int q = 0; q < 8; ++q) dst[base + q*L] = { p[q], r[q] };
    }
    f4* t = src; src = dst; dst = t;
    L <<= 3;
  }
  __syncthreads();
  return src;
}

// ---------------------------------------------------------------------------
// Spectral stage for one (k, N-k) pair of one sequence:
// unpack X from packed Z, Y = A*X, repack z' for the inverse packed FFT.
// ---------------------------------------------------------------------------
__device__ inline void spec_pair(c2 Zk, c2 Zm, c2 Ak, c2 Am,
                                 float cw, float sw, c2& zk, c2& zm)
{
  // E=(Zk+conj(Zm))/2, D=(Zk-conj(Zm))/2, O=-i*D, X_k=E+W*O, W=e^{-i pi k/N}
  c2 E  = {0.5f*(Zk.x + Zm.x), 0.5f*(Zk.y - Zm.y)};
  c2 Dd = {0.5f*(Zk.x - Zm.x), 0.5f*(Zk.y + Zm.y)};
  c2 O  = {Dd.y, -Dd.x};
  c2 W1 = {cw, sw};
  c2 WO = cmul(W1, O);
  c2 Xk = {E.x + WO.x, E.y + WO.y};
  c2 Xm = {E.x - WO.x, -(E.y - WO.y)};      // X[N-k] = conj(E - W*O)
  c2 Yk = cmul(Ak, Xk);
  c2 Ym = cmul(Am, Xm);
  // z'[k] = E' + i*O', E'=(Yk+conj(Ym))/2, O'=0.5*V*(Yk-conj(Ym)), V=e^{+i pi k/N}
  c2 Ep = {0.5f*(Yk.x + Ym.x), 0.5f*(Yk.y - Ym.y)};
  c2 Dp = {0.5f*(Yk.x - Ym.x), 0.5f*(Yk.y + Ym.y)};
  c2 V  = {cw, -sw};
  c2 Op = cmul(V, Dp);
  zk = {Ep.x - Op.y, Ep.y + Op.x};
  // z'[N-k]: V^{N-k} = -e^{-i pi k/N} = {-cw, -sw}
  c2 E2 = {0.5f*(Ym.x + Yk.x), 0.5f*(Ym.y - Yk.y)};
  c2 D2 = {0.5f*(Ym.x - Yk.x), 0.5f*(Ym.y + Yk.y)};
  c2 V2 = {-cw, -sw};
  c2 O2 = cmul(V2, D2);
  zm = {E2.x - O2.y, E2.y + O2.x};
}

// ---------------------------------------------------------------------------
// Kernel 2: fused rfft(8192) -> spectral multiply -> irfft.
// One WG per (b, h, d-pair): two adjacent channels share twiddles and ride in
// 16-byte LDS elements (ds_*_b128) and 8-byte global accesses.
// ---------------------------------------------------------------------------
__global__ void __launch_bounds__(256)
fftconv_kernel(const float* __restrict__ x, const float* __restrict__ filt,
               float* __restrict__ out)
{
  __shared__ f4 sP[4096];                    // 64 KB
  __shared__ f4 sQ[4096];                    // 64 KB  (128 KB/WG, <= 320 KB)
  const int tid = threadIdx.x;
  const int id  = blockIdx.x;
  const int dp = id & 31;                    // d-pair
  const int h  = (id >> 5) & 7;
  const int b  = id >> 8;
  const int d0 = dp * 2;

  const size_t xoff = ((size_t)(b*HH + h) * NSEQ) * DD + d0;
  const float* xp = x + xoff;
  const float* Fh = filt + (size_t)h * NF * 128 + d0;

  // pack z[j] = x[2j] + i*x[2j+1] (zero-padded 8192-pt real signal), both chans
  for(int j = tid; j < 2048; j += 256){
    float2 e = *(const float2*)(xp + (size_t)(2*j)   * DD);
    float2 o = *(const float2*)(xp + (size_t)(2*j+1) * DD);
    f4 z; z.s0 = {e.x, o.x}; z.s1 = {e.y, o.y};
    sP[j] = z;
    f4 zz; zz.s0 = {0.f,0.f}; zz.s1 = {0.f,0.f};
    sP[j + 2048] = zz;
    __builtin_prefetch(Fh + (size_t)(2*j) * 128, 0, 1);   // warm filter in L2
  }

  f4* Z = fft4096_r8<-1>(sP, sQ, tid);       // forward -> sP

  for(int k = tid; k <= 2048; k += 256){
    if(k == 0){
      f4 Z0 = Z[0];
      float2 A0 = *(const float2*)(Fh);
      float2 AN = *(const float2*)(Fh + (size_t)4096*128);
      f4 zq;
      { float X0 = Z0.s0.x + Z0.s0.y, XN = Z0.s0.x - Z0.s0.y;
        float Y0 = A0.x * X0, YN = AN.x * XN;
        zq.s0 = {0.5f*(Y0 + YN), 0.5f*(Y0 - YN)}; }
      { float X0 = Z0.s1.x + Z0.s1.y, XN = Z0.s1.x - Z0.s1.y;
        float Y0 = A0.y * X0, YN = AN.y * XN;
        zq.s1 = {0.5f*(Y0 + YN), 0.5f*(Y0 - YN)}; }
      sQ[0] = zq;
    } else {
      int km = 4096 - k;                     // k=2048 self-paired
      f4 Zk = Z[k], Zm = Z[km];
      float ang = -PI_F * (float)k * (1.0f/4096.0f);
      float sw, cw; __sincosf(ang, &sw, &cw);
      float2 Akre = *(const float2*)(Fh + (size_t)k *128);
      float2 Akim = *(const float2*)(Fh + (size_t)k *128 + 64);
      float2 Amre = *(const float2*)(Fh + (size_t)km*128);
      float2 Amim = *(const float2*)(Fh + (size_t)km*128 + 64);
      c2 zk0, zm0, zk1, zm1;
      spec_pair(Zk.s0, Zm.s0, {Akre.x, Akim.x}, {Amre.x, Amim.x}, cw, sw, zk0, zm0);
      spec_pair(Zk.s1, Zm.s1, {Akre.y, Akim.y}, {Amre.y, Amim.y}, cw, sw, zk1, zm1);
      f4 qk; qk.s0 = zk0; qk.s1 = zk1;
      sQ[k] = qk;
      if(km != k){
        f4 qm; qm.s0 = zm0; qm.s1 = zm1;
        sQ[km] = qm;
      }
    }
  }

  f4* R = fft4096_r8<+1>(sQ, sP, tid);       // inverse (unnormalized) -> sQ

  float* op = out + xoff;
  const float sc = 1.0f/4096.0f;
  for(int j = tid; j < 2048; j += 256){      // first 4096 of 8192 samples
    f4 z = R[j];
    float2 ve; ve.x = z.s0.x * sc; ve.y = z.s1.x * sc;
    float2 vo; vo.x = z.s0.y * sc; vo.y = z.s1.y * sc;
    *(float2*)(op + (size_t)(2*j)   * DD) = ve;
    *(float2*)(op + (size_t)(2*j+1) * DD) = vo;
  }
}

// ---------------------------------------------------------------------------
extern "C" void kernel_launch(void* const* d_in, const int* in_sizes, int n_in,
                              void* d_out, int out_size, void* d_ws, size_t ws_size,
                              hipStream_t stream)
{
  (void)in_sizes; (void)n_in; (void)out_size; (void)ws_size;
  const float* x        = (const float*)d_in[0];
  const float* w_in     = (const float*)d_in[1];
  const float* b_in     = (const float*)d_in[2];
  const float* w_layers = (const float*)d_in[3];
  const float* b_layers = (const float*)d_in[4];
  const float* w_out    = (const float*)d_in[5];
  const float* b_out    = (const float*)d_in[6];
  float* filt = (float*)d_ws;                // 8*4097*128 floats = 16.78 MB

  rpe_mlp_kernel<<<dim3((NF + 15) / 16), dim3(32), 0, stream>>>(
      w_in, b_in, w_layers, b_layers, w_out, b_out, filt);

  fftconv_kernel<<<dim3(BB * HH * DD / 2), dim3(256), 0, stream>>>(
      x, filt, (float*)d_out);
}
